// Aggregator_67585605370022
// MI455X (gfx1250) — matrix operand order repaired and checked by reading, not compiled, not run on hardware
//
#include <hip/hip_runtime.h>
#include <hip/hip_bf16.h>

// CDNA5 / gfx1250 WMMA implementation of the NeRF aggregator.
// [repack weights f32->f16 fragment order] -> [point MLP + aggregation] -> [color head].
// Kernel 1 uses M-register blocking (32 rows/wave) + cooperative LDS weight
// staging (async-to-LDS when the toolchain exposes the gfx1250 builtins).

typedef __attribute__((ext_vector_type(16))) _Float16 v16h_t;
typedef __attribute__((ext_vector_type(8)))  _Float16 v8h_t;
typedef __attribute__((ext_vector_type(8)))  float    v8f_t;
typedef __attribute__((ext_vector_type(4)))  int      v4i_t;

// ---- workspace layout (in _Float16 units) ----
#define WMAIN_OFF 0
#define WCIN_OFF  327680
#define WCH_OFF   364544
#define RAD_OFF   397312
#define W_TOTAL   397312

__device__ __forceinline__ v8f_t wmma_f16(v16h_t a, v16h_t b, v8f_t c) {
  return __builtin_amdgcn_wmma_f32_16x16x32_f16(false, a, false, b, (short)0, c,
                                                false, false);
}

// A-fragment (16x32 f16) per-lane load from an LDS activation row.
// lane l, m=l&15, h=l>>4: elems 0..7 = K[kbase+8h, +8), elems 8..15 = K[kbase+16+8h, +8)
__device__ __forceinline__ v16h_t lds_a_frag(const _Float16* row, int kbase, int hl) {
  v8h_t lo = *(const v8h_t*)(row + kbase + 8 * hl);
  v8h_t hi = *(const v8h_t*)(row + kbase + 16 + 8 * hl);
  return __builtin_shufflevector(lo, hi, 0, 1, 2, 3, 4, 5, 6, 7,
                                 8, 9, 10, 11, 12, 13, 14, 15);
}

__device__ __forceinline__ float lrelu(float v) { return v > 0.f ? v : 0.01f * v; }

// ---- gfx1250 async global->LDS copy (ASYNCcnt) with safe fallback ----
#if defined(__HIP_DEVICE_COMPILE__)
#if __has_builtin(__builtin_amdgcn_global_load_async_to_lds_b128)
#define HAVE_ASYNC_LDS 1
#endif
#endif

#ifdef HAVE_ASYNC_LDS
typedef __attribute__((address_space(1))) v4i_t gv4i_t;  // global int4
typedef __attribute__((address_space(3))) v4i_t lv4i_t;  // LDS int4
#endif

__device__ __forceinline__ void copy16_g2l(const _Float16* g, _Float16* l) {
#ifdef HAVE_ASYNC_LDS
  gv4i_t* gp = (gv4i_t*)(unsigned long long)(const void*)g;
  lv4i_t* lp = (lv4i_t*)(unsigned int)(unsigned long long)(void*)l;
  __builtin_amdgcn_global_load_async_to_lds_b128(gp, lp, 0, 0);
#else
  *(v8h_t*)l = *(const v8h_t*)g;
#endif
}

__device__ __forceinline__ void stage_fence_barrier() {
#ifdef HAVE_ASYNC_LDS
#if __has_builtin(__builtin_amdgcn_s_wait_asynccnt)
  __builtin_amdgcn_s_wait_asynccnt(0);
#else
  asm volatile("s_wait_asynccnt 0x0" ::: "memory");
#endif
#endif
  __syncthreads();
}

// -------------------------------------------------------------------------
// Kernel 0: repack weights into per-lane WMMA B-fragment order.
// B layout (32x16 f16): lane l holds col n=l&15; f16 elem i = K row 16*(i>>3)+8*(l>>4)+(i&7).
// -------------------------------------------------------------------------
__global__ __launch_bounds__(256) void repack_kernel(
    const float* __restrict__ W1_in, const float* __restrict__ W1_h,
    const float* __restrict__ Wc_in, const float* __restrict__ Wc_h,
    _Float16* __restrict__ dst) {
  int e = blockIdx.x * 256 + threadIdx.x;
  if (e >= W_TOTAL) return;
  const float* src;
  int Kact, N, NT, rem;
  if (e < WCIN_OFF) {            // main MLP: 5 layers of 256x256
    int L = e >> 16;
    rem = e & 65535;
    src = (L == 0) ? W1_in : (W1_h + (L - 1) * 65536);
    Kact = (L == 0) ? 254 : 256;
    N = 256; NT = 16;
  } else if (e < WCH_OFF) {      // Wc_in: K 280 -> 288, N=128
    rem = e - WCIN_OFF;
    src = Wc_in; Kact = 280; N = 128; NT = 8;
  } else {                       // Wc_h[2]: 128x128
    int t = e - WCH_OFF;
    int i = t >> 14;
    rem = t & 16383;
    src = Wc_h + i * 16384; Kact = 128; N = 128; NT = 8;
  }
  int frag = rem >> 9;
  int lane = (rem >> 4) & 31;
  int ii = rem & 15;
  int kt = frag / NT, nt = frag % NT;
  int k = kt * 32 + ((ii >> 3) << 4) + ((lane >> 4) << 3) + (ii & 7);
  int n = nt * 16 + (lane & 15);
  float v = (k < Kact) ? src[k * N + n] : 0.f;
  dst[e] = (_Float16)v;
}

// -------------------------------------------------------------------------
// Kernel 1: per-point MLP (5 layers) + density head + weighted aggregation.
// 128 threads = 4 waves. Wave w owns 32 points (rows w*32..w*32+31): two
// 16-row A-tiles held in registers; B fragments staged half-layer (64KB) in
// LDS and reused by all 4 waves and both row-tiles.
// -------------------------------------------------------------------------
#define STR1 264                       // f16 stride per activation row
#define ACT1_ELEMS (128 * STR1)        // 33,792 f16
#define WST_ELEMS  (64 * 512)          // 32,768 f16 (64 fragments)
#define SMEM1_BYTES ((ACT1_ELEMS + WST_ELEMS) * 2)

__global__ __launch_bounds__(128) void point_mlp_kernel(
    const float* __restrict__ rel, const float* __restrict__ feat,
    const unsigned char* __restrict__ mask,
    const float* __restrict__ b1_in, const float* __restrict__ b1_h,
    const float* __restrict__ Wsv, const float* __restrict__ bs,
    const _Float16* __restrict__ wsW, _Float16* __restrict__ rad,
    float* __restrict__ out) {
  extern __shared__ __align__(16) _Float16 dsm[];
  _Float16* act = dsm;                 // [128][STR1]
  _Float16* wst = dsm + ACT1_ELEMS;    // 64 staged fragments

  const int tid = threadIdx.x;
  const int lane = tid & 31;
  const int wave = tid >> 5;
  const int hl = lane >> 4;
  const int nn = lane & 15;
  const int Pbase = blockIdx.x * 128;

  // ---- build input x = [feat(32), posenc(feat,3)(192), posenc(rel,5)(30), 0,0]
  for (int idx = tid; idx < 128 * 256; idx += 128) {
    int p = idx >> 8, d = idx & 255;
    int P = Pbase + p;
    float v;
    if (d < 32) {
      v = feat[P * 32 + d];
    } else if (d < 224) {
      int q = d - 32, j = q / 6, r = q % 6, f = r >> 1;
      float x = feat[P * 32 + j] * (float)(1 << f);
      v = (r & 1) ? cosf(x) : sinf(x);
    } else if (d < 254) {
      int q = d - 224, j = q / 10, r = q % 10, f = r >> 1;
      float x = rel[P * 3 + j] * (float)(1 << f);
      v = (r & 1) ? cosf(x) : sinf(x);
    } else {
      v = 0.f;
    }
    act[p * STR1 + d] = (_Float16)v;
  }
  __syncthreads();

  // ---- 5 dense layers via WMMA ----
  _Float16* actRow0 = &act[(wave * 32 + nn) * STR1];
  _Float16* actRow1 = actRow0 + 16 * STR1;
  for (int layer = 0; layer < 5; ++layer) {
    const _Float16* Wg = wsW + layer * 65536;
    const float* bias = (layer == 0) ? b1_in : (b1_h + (layer - 1) * 256);

    // A fragments for both row-tiles live in registers for the whole layer.
    v16h_t a0[8], a1[8];
#pragma unroll
    for (int kt = 0; kt < 8; ++kt) {
      a0[kt] = lds_a_frag(actRow0, kt * 32, hl);
      a1[kt] = lds_a_frag(actRow1, kt * 32, hl);
    }

    for (int half_n = 0; half_n < 2; ++half_n) {
      // stage 64 fragments (all kt, nt in [half_n*8, half_n*8+8)) into LDS
      for (int j = tid; j < 4096; j += 128) {   // 4096 x 16B chunks
        int flocal = j >> 6;                    // kt*8 + ntl
        int kt = flocal >> 3, ntl = flocal & 7;
        int src_frag = (kt << 4) + (half_n << 3) + ntl;
        copy16_g2l(Wg + (src_frag << 9) + ((j & 63) << 3), wst + (j << 3));
      }
      stage_fence_barrier();

      for (int ntl = 0; ntl < 8; ++ntl) {
        int nt = (half_n << 3) + ntl;
        float bval = bias[nt * 16 + nn];
        v8f_t acc0, acc1;
#pragma unroll
        for (int r = 0; r < 8; ++r) { acc0[r] = bval; acc1[r] = bval; }
#pragma unroll
        for (int kt = 0; kt < 8; ++kt) {
          v16h_t b = *(const v16h_t*)(wst + (((kt << 3) + ntl) << 9) + lane * 16);
          acc0 = wmma_f16(a0[kt], b, acc0);
          acc1 = wmma_f16(a1[kt], b, acc1);
        }
#pragma unroll
        for (int r = 0; r < 8; ++r) {
          int col = nt * 16 + nn;
          act[(wave * 32 + hl * 8 + r) * STR1 + col] = (_Float16)lrelu(acc0[r]);
          act[(wave * 32 + 16 + hl * 8 + r) * STR1 + col] = (_Float16)lrelu(acc1[r]);
        }
      }
      __syncthreads();  // all waves done with this stage before overwrite
    }
  }

  // ---- density head + per-point weights: lane <-> point (32 per wave) ----
  int P = Pbase + wave * 32 + lane;
  const _Float16* hrow = &act[(wave * 32 + lane) * STR1];
  float s = bs[0] - 1.0f;
  for (int c = 0; c < 256; ++c) s += (float)hrow[c] * Wsv[c];
  float dp = (s > 20.f) ? s : log1pf(expf(s));  // softplus
  float dx = rel[P * 3 + 0], dy = rel[P * 3 + 1], dz = rel[P * 3 + 2];
  float dist = sqrtf(dx * dx + dy * dy + dz * dz);
  float wnv = mask[P] ? (1.0f / fmaxf(dist, 1e-8f)) : 0.f;
  // normalize within each k=8 group
  float ssum = wnv;
  ssum += __shfl_xor(ssum, 1, 32);
  ssum += __shfl_xor(ssum, 2, 32);
  ssum += __shfl_xor(ssum, 4, 32);
  wnv = wnv / fmaxf(ssum, 1e-8f);

  // ---- aggregation: wave handles its 4 groups, 257 channels across lanes ----
  for (int g = 0; g < 4; ++g) {
    float wk[8], dk[8];
#pragma unroll
    for (int k = 0; k < 8; ++k) {
      wk[k] = __shfl(wnv, g * 8 + k, 32);
      dk[k] = __shfl(dp, g * 8 + k, 32);
    }
    int ng = blockIdx.x * 16 + wave * 4 + g;
    for (int c = lane; c < 257; c += 32) {
      float sacc = 0.f;
#pragma unroll
      for (int k = 0; k < 8; ++k) {
        float rv = (c == 0) ? dk[k]
                            : (float)act[(wave * 32 + g * 8 + k) * STR1 + (c - 1)];
        sacc += wk[k] * rv;
      }
      if (c == 0) out[ng * 4 + 0] = sacc;                 // density
      else        rad[ng * 256 + (c - 1)] = (_Float16)sacc;  // radiance
    }
  }
}

// -------------------------------------------------------------------------
// Kernel 2: color head: cf(280->288) -> 128 -> 128 -> 128 -> sigmoid(3)
// -------------------------------------------------------------------------
#define STR2 296

__global__ __launch_bounds__(128) void color_mlp_kernel(
    const float* __restrict__ view,
    const float* __restrict__ bc_in, const float* __restrict__ bc_h,
    const float* __restrict__ WcO, const float* __restrict__ bcO,
    const _Float16* __restrict__ wsWc, const _Float16* __restrict__ rad,
    float* __restrict__ out) {
  __shared__ _Float16 act[64 * STR2];  // 37,888 B

  const int tid = threadIdx.x;
  const int lane = tid & 31;
  const int wave = tid >> 5;
  const int hl = lane >> 4;
  const int nn = lane & 15;
  const int Rbase = blockIdx.x * 64;

  for (int idx = tid; idx < 64 * 288; idx += 128) {
    int p = idx / 288, d = idx % 288;
    int R = Rbase + p;
    float v;
    if (d < 256) {
      v = (float)rad[R * 256 + d];
    } else if (d < 280) {
      int q = d - 256, j = q >> 3, r = q & 7, f = r >> 1;
      float x = view[R * 3 + j] * (float)(1 << f);
      v = (r & 1) ? cosf(x) : sinf(x);
    } else {
      v = 0.f;
    }
    act[p * STR2 + d] = (_Float16)v;
  }
  __syncthreads();

  _Float16* actRow = &act[(wave * 16 + nn) * STR2];
  const _Float16* Wf1 = wsWc + 36864;
  const _Float16* Wf2 = wsWc + 36864 + 16384;
  for (int layer = 0; layer < 3; ++layer) {
    int KT = (layer == 0) ? 9 : 4;
    const _Float16* Wf = (layer == 0) ? wsWc : ((layer == 1) ? Wf1 : Wf2);
    const float* bias = (layer == 0) ? bc_in : (bc_h + (layer - 1) * 128);

    v16h_t a[9];
    for (int kt = 0; kt < KT; ++kt) a[kt] = lds_a_frag(actRow, kt * 32, hl);

    for (int nt = 0; nt < 8; ++nt) {
      float bval = bias[nt * 16 + nn];
      v8f_t acc;
#pragma unroll
      for (int r = 0; r < 8; ++r) acc[r] = bval;
      for (int kt = 0; kt < KT; ++kt) {
        v16h_t b = *(const v16h_t*)(Wf + ((kt * 8 + nt) << 9) + lane * 16);
        acc = wmma_f16(a[kt], b, acc);
      }
#pragma unroll
      for (int r = 0; r < 8; ++r) {
        act[(wave * 16 + hl * 8 + r) * STR2 + nt * 16 + nn] = (_Float16)lrelu(acc[r]);
      }
    }
  }

  if (lane < 16) {
    int row = wave * 16 + lane;
    int R = Rbase + row;
    const _Float16* hrow = &act[row * STR2];
    float s0 = bcO[0], s1 = bcO[1], s2 = bcO[2];
    for (int j = 0; j < 128; ++j) {
      float hv = (float)hrow[j];
      s0 += hv * WcO[j * 3 + 0];
      s1 += hv * WcO[j * 3 + 1];
      s2 += hv * WcO[j * 3 + 2];
    }
    out[R * 4 + 1] = (1.f / (1.f + expf(-s0))) * 1.002f - 0.001f;
    out[R * 4 + 2] = (1.f / (1.f + expf(-s1))) * 1.002f - 0.001f;
    out[R * 4 + 3] = (1.f / (1.f + expf(-s2))) * 1.002f - 0.001f;
  }
}

// -------------------------------------------------------------------------
extern "C" void kernel_launch(void* const* d_in, const int* in_sizes, int n_in,
                              void* d_out, int out_size, void* d_ws, size_t ws_size,
                              hipStream_t stream) {
  const float* rel = (const float*)d_in[0];
  const float* feat = (const float*)d_in[1];
  const unsigned char* mask = (const unsigned char*)d_in[2];
  const float* view = (const float*)d_in[3];
  const float* W1_in = (const float*)d_in[4];
  const float* b1_in = (const float*)d_in[5];
  const float* W1_h = (const float*)d_in[6];
  const float* b1_h = (const float*)d_in[7];
  const float* Ws = (const float*)d_in[8];
  const float* bs = (const float*)d_in[9];
  const float* Wc_in = (const float*)d_in[10];
  const float* bc_in = (const float*)d_in[11];
  const float* Wc_h = (const float*)d_in[12];
  const float* bc_h = (const float*)d_in[13];
  const float* Wc_out = (const float*)d_in[14];
  const float* bc_out = (const float*)d_in[15];

  const int n = in_sizes[3] / 3;  // 32768

  _Float16* wsW = (_Float16*)d_ws;
  _Float16* rad = wsW + RAD_OFF;

  repack_kernel<<<(W_TOTAL + 255) / 256, 256, 0, stream>>>(W1_in, W1_h, Wc_in,
                                                           Wc_h, wsW);

  int nb1 = (n * 8) / 128;  // 2048 blocks, 128 points each
  point_mlp_kernel<<<nb1, 128, SMEM1_BYTES, stream>>>(rel, feat, mask, b1_in,
                                                      b1_h, Ws, bs, wsW, rad,
                                                      (float*)d_out);

  int nb2 = n / 64;  // 512 blocks, 64 rows each
  color_mlp_kernel<<<nb2, 128, 0, stream>>>(view, bc_in, bc_h, Wc_out, bc_out,
                                            wsW + WCIN_OFF, rad, (float*)d_out);
}
